// Encoder_19619410608214
// MI455X (gfx1250) — matrix operand order repaired and checked by reading, not compiled, not run on hardware
//
#include <hip/hip_runtime.h>
#include <hip/hip_bf16.h>
#include <math.h>

#define RGAT_R 8
#define RGAT_H 64
#define RGAT_OUT 32
#define NEG_SLOPE 0.2f

typedef __attribute__((ext_vector_type(16))) __bf16 bf16x16;
typedef __attribute__((ext_vector_type(8)))  float  f32x8;

// ---------------------------------------------------------------------------
// Init per-layer scratch: m = -inf, s = 0, acc = 0
// ---------------------------------------------------------------------------
__global__ void init_buffers(float* __restrict__ m, float* __restrict__ s,
                             float* __restrict__ acc, int n_nodes) {
    int i = blockIdx.x * blockDim.x + threadIdx.x;
    if (i < n_nodes) { m[i] = -INFINITY; s[i] = 0.f; }
    if (i < n_nodes * RGAT_H) acc[i] = 0.f;
}

// ---------------------------------------------------------------------------
// Pack W (R x K x 64, f32) into per-lane WMMA B-fragments (bf16):
//   Wp[((r*kblocks + kb)*4 + tile_o)*32 + lane][e] ,  e=0..15
//   element e of lane -> K = kb*32 + 8*(lane>=16) + e + 8*(e>=8), col = tile_o*16 + lane%16
// One thread per fragment (r, kb, tile_o, lane).
// ---------------------------------------------------------------------------
__global__ void pack_weights(const float* __restrict__ W, __bf16* __restrict__ Wp,
                             int K) {
    const int kblocks = K >> 5;
    const int total = RGAT_R * kblocks * 4 * 32;
    int idx = blockIdx.x * blockDim.x + threadIdx.x;
    if (idx >= total) return;
    const int lane   = idx & 31;
    const int tile_o = (idx >> 5) & 3;
    const int kb     = (idx >> 7) % kblocks;
    const int r      = (idx >> 7) / kblocks;
    const int half = lane >> 4;
    const int col  = tile_o * 16 + (lane & 15);
    const float* __restrict__ wsrc = W + (size_t)r * K * RGAT_H;
    __bf16* __restrict__ dstp = Wp + (size_t)idx * 16;
#pragma unroll
    for (int e = 0; e < 16; ++e) {
        int k = kb * 32 + 8 * half + e + ((e >> 3) << 3);
        dstp[e] = (__bf16)wsrc[(size_t)k * RGAT_H + col];
    }
}

// ---------------------------------------------------------------------------
// T[r] = A(n_nodes x K) @ W[r](K x 64).  One wave computes a 16x64 strip:
// 4 accumulators share one A fragment per 32-wide K step (4 WMMAs / A fetch).
// A loaded as float4 (global_load_b128), B as contiguous 32B bf16 fragments.
// Requires n_nodes % 16 == 0, K % 32 == 0 (holds: 50000, 128/64).
// ---------------------------------------------------------------------------
__global__ void rgat_gemm_wmma(const float* __restrict__ A,
                               const __bf16* __restrict__ Wp,
                               float* __restrict__ T,
                               int n_nodes, int K) {
    const int gtid = blockIdx.x * blockDim.x + threadIdx.x;
    const int wave = gtid >> 5;
    const int lane = threadIdx.x & 31;
    const int tiles_m = n_nodes >> 4;
    const int r = wave / tiles_m;                 // wave-uniform guard
    if (r >= RGAT_R) return;
    const int tile_m = wave - r * tiles_m;
    const int half = lane >> 4;
    const int lr   = lane & 15;

    const int m_row = tile_m * 16 + lr;
    const float* __restrict__ arow = A + (size_t)m_row * K;
    const int kblocks = K >> 5;
    const __bf16* __restrict__ wlane =
        Wp + ((size_t)r * kblocks * 4 * 32 + lane) * 16;

    f32x8 c0 = {}, c1 = {}, c2 = {}, c3 = {};
    for (int kb = 0; kb < kblocks; ++kb) {
        // A fragment (ISA 16-bit A layout): e<8 -> K=e+8*half, e>=8 -> K=16+e-8+8*half
        const float* ap = arow + kb * 32 + half * 8;
        float4 x0 = ((const float4*)ap)[0];
        float4 x1 = ((const float4*)ap)[1];
        float4 x2 = ((const float4*)(ap + 16))[0];
        float4 x3 = ((const float4*)(ap + 16))[1];
        bf16x16 a;
        a[0]=(__bf16)x0.x; a[1]=(__bf16)x0.y; a[2]=(__bf16)x0.z; a[3]=(__bf16)x0.w;
        a[4]=(__bf16)x1.x; a[5]=(__bf16)x1.y; a[6]=(__bf16)x1.z; a[7]=(__bf16)x1.w;
        a[8]=(__bf16)x2.x; a[9]=(__bf16)x2.y; a[10]=(__bf16)x2.z; a[11]=(__bf16)x2.w;
        a[12]=(__bf16)x3.x; a[13]=(__bf16)x3.y; a[14]=(__bf16)x3.z; a[15]=(__bf16)x3.w;

        const __bf16* wkb = wlane + (size_t)kb * (4 * 32 * 16);
        bf16x16 b0 = *(const bf16x16*)(wkb);
        bf16x16 b1 = *(const bf16x16*)(wkb + 32 * 16);
        bf16x16 b2 = *(const bf16x16*)(wkb + 2 * 32 * 16);
        bf16x16 b3 = *(const bf16x16*)(wkb + 3 * 32 * 16);

        c0 = __builtin_amdgcn_wmma_f32_16x16x32_bf16(false, a, false, b0, (short)0, c0, false, false);
        c1 = __builtin_amdgcn_wmma_f32_16x16x32_bf16(false, a, false, b1, (short)0, c1, false, false);
        c2 = __builtin_amdgcn_wmma_f32_16x16x32_bf16(false, a, false, b2, (short)0, c2, false, false);
        c3 = __builtin_amdgcn_wmma_f32_16x16x32_bf16(false, a, false, b3, (short)0, c3, false, false);
    }
    // C layout: VGPR v -> row v + 8*half, lane lr -> column within each 16-wide tile
    float* __restrict__ trow = T + (size_t)r * n_nodes * RGAT_H;
#pragma unroll
    for (int v = 0; v < 8; ++v) {
        int row = tile_m * 16 + v + half * 8;
        float* p = trow + (size_t)row * RGAT_H + lr;
        p[0]  = c0[v];
        p[16] = c1[v];
        p[32] = c2[v];
        p[48] = c3[v];
    }
}

// ---------------------------------------------------------------------------
// dq[r,n] = T[r,n,:]·q ; dk[r,n] = T[r,n,:]·k  (one wave per (r,n) row)
// ---------------------------------------------------------------------------
__global__ void qk_dot(const float* __restrict__ T,
                       const float* __restrict__ q, const float* __restrict__ k,
                       float* __restrict__ dq, float* __restrict__ dk,
                       int n_rows) {
    int wave = (blockIdx.x * blockDim.x + threadIdx.x) >> 5;
    int lane = threadIdx.x & 31;
    if (wave >= n_rows) return;
    const float* row = T + (size_t)wave * RGAT_H;
    float v0 = row[lane], v1 = row[lane + 32];
    float pq = v0 * q[lane] + v1 * q[lane + 32];
    float pk = v0 * k[lane] + v1 * k[lane + 32];
#pragma unroll
    for (int off = 16; off > 0; off >>= 1) {
        pq += __shfl_down(pq, off, 32);
        pk += __shfl_down(pk, off, 32);
    }
    if (lane == 0) { dq[wave] = pq; dk[wave] = pk; }
}

// ---------------------------------------------------------------------------
// float atomic max via sign-split int/uint atomics (m initialized to -inf)
// ---------------------------------------------------------------------------
__device__ __forceinline__ void atomicMaxF(float* addr, float val) {
    if (val >= 0.f) atomicMax((int*)addr, __float_as_int(val));
    else            atomicMin((unsigned int*)addr, __float_as_uint(val));
}

__device__ __forceinline__ float leaky(float a) {
    return a >= 0.f ? a : NEG_SLOPE * a;
}

// Pass 1: logits -> store to ex[], atomic segment max into m[]
__global__ void edge_logit_max(const int* __restrict__ src, const int* __restrict__ dst,
                               const int* __restrict__ et,
                               const float* __restrict__ dq, const float* __restrict__ dk,
                               float* __restrict__ m, float* __restrict__ ex,
                               int E, int n_nodes) {
    int e = blockIdx.x * blockDim.x + threadIdx.x;
    if (e >= E) return;
    int s = src[e], d = dst[e], r = et[e];
    float a = leaky(dq[(size_t)r * n_nodes + d] + dk[(size_t)r * n_nodes + s]);
    ex[e] = a;
    atomicMaxF(&m[d], a);
}

// Pass 2: ex[e] = exp(a - m[dst]); atomic segment sum into ssum[]
__global__ void edge_exp_sum(const int* __restrict__ dst,
                             const float* __restrict__ m, float* __restrict__ ex,
                             float* __restrict__ ssum, int E) {
    int e = blockIdx.x * blockDim.x + threadIdx.x;
    if (e >= E) return;
    int d = dst[e];
    float v = expf(ex[e] - m[d]);
    ex[e] = v;
    atomicAdd(&ssum[d], v);
}

// Pass 3: one wave per edge, lanes cover the 64-wide row (2 floats/lane):
// coalesced 128B gathers from L2-resident T, coalesced f32 atomics into acc.
__global__ void edge_aggregate(const int* __restrict__ src, const int* __restrict__ dst,
                               const int* __restrict__ et, const float* __restrict__ T,
                               const float* __restrict__ ex, const float* __restrict__ ssum,
                               float* __restrict__ acc, int E, int n_nodes) {
    int wave = (blockIdx.x * blockDim.x + threadIdx.x) >> 5;
    int lane = threadIdx.x & 31;
    if (wave >= E) return;
    int s = src[wave], d = dst[wave], r = et[wave];
    float alpha = ex[wave] / (ssum[d] + 1e-16f);
    const float* row = T + ((size_t)r * n_nodes + s) * RGAT_H;
    atomicAdd(&acc[(size_t)d * RGAT_H + lane],      alpha * row[lane]);
    atomicAdd(&acc[(size_t)d * RGAT_H + lane + 32], alpha * row[lane + 32]);
}

__global__ void bias_relu(const float* __restrict__ acc, const float* __restrict__ bias,
                          float* __restrict__ outh, int n_nodes) {
    int i = blockIdx.x * blockDim.x + threadIdx.x;
    if (i >= n_nodes * RGAT_H) return;
    float v = acc[i] + bias[i & (RGAT_H - 1)];
    outh[i] = v > 0.f ? v : 0.f;
}

// out[n,o] = scale * (H[n,:] @ W[:,o] + b[o]);  scale folds the 0.5 for logstd
__global__ void final_linear(const float* __restrict__ Hbuf, const float* __restrict__ Wt,
                             const float* __restrict__ bias, float* __restrict__ out,
                             float scale, int n_nodes) {
    int idx = blockIdx.x * blockDim.x + threadIdx.x;
    if (idx >= n_nodes * RGAT_OUT) return;
    int n = idx >> 5, o = idx & (RGAT_OUT - 1);
    const float* hrow = Hbuf + (size_t)n * RGAT_H;
    float a = bias[o];
#pragma unroll
    for (int kk = 0; kk < RGAT_H; ++kk) a = fmaf(hrow[kk], Wt[kk * RGAT_OUT + o], a);
    out[idx] = scale * a;
}

// ---------------------------------------------------------------------------
extern "C" void kernel_launch(void* const* d_in, const int* in_sizes, int n_in,
                              void* d_out, int out_size, void* d_ws, size_t ws_size,
                              hipStream_t stream) {
    (void)n_in; (void)out_size; (void)ws_size;
    const float* x   = (const float*)d_in[0];
    const int*   ei  = (const int*)d_in[1];
    const int*   et  = (const int*)d_in[2];
    const float* W1  = (const float*)d_in[3];
    const float* q1  = (const float*)d_in[4];
    const float* k1  = (const float*)d_in[5];
    const float* b1  = (const float*)d_in[6];
    const float* Wmu = (const float*)d_in[7];
    const float* qmu = (const float*)d_in[8];
    const float* kmu = (const float*)d_in[9];
    const float* bmu = (const float*)d_in[10];
    const float* Wlv = (const float*)d_in[11];
    const float* qlv = (const float*)d_in[12];
    const float* klv = (const float*)d_in[13];
    const float* blv = (const float*)d_in[14];
    const float* Wm  = (const float*)d_in[15];
    const float* bm  = (const float*)d_in[16];
    const float* Wl  = (const float*)d_in[17];
    const float* bl  = (const float*)d_in[18];

    const int n_nodes = in_sizes[0] / 128;   // IN = 128
    const int E       = in_sizes[1] / 2;
    const int* src = ei;
    const int* dst = ei + E;

    float* ws = (float*)d_ws;
    size_t off = 0;
    float* T    = ws + off; off += (size_t)RGAT_R * n_nodes * RGAT_H;  // 102.4 MB
    float* dq   = ws + off; off += (size_t)RGAT_R * n_nodes;
    float* dk   = ws + off; off += (size_t)RGAT_R * n_nodes;
    float* mmax = ws + off; off += (size_t)n_nodes;
    float* ssum = ws + off; off += (size_t)n_nodes;
    float* ex   = ws + off; off += (size_t)E;
    float* acc  = ws + off; off += (size_t)n_nodes * RGAT_H;
    float* hid  = ws + off; off += (size_t)n_nodes * RGAT_H;
    float* h2   = ws + off; off += (size_t)n_nodes * RGAT_H;
    off = (off + 15) & ~(size_t)15;                       // 32B-align Wp
    __bf16* Wp  = (__bf16*)(ws + off);                    // up to R*128*64 bf16 (128 KB)

    const int eb   = (E + 255) / 256;                     // thread-per-edge
    const int ewb  = (E + 7) / 8;                         // wave-per-edge
    const int nb64 = (n_nodes * RGAT_H + 255) / 256;
    const int qkb  = ((RGAT_R * n_nodes) * 32 + 255) / 256;

    auto run_layer = [&](const float* Act, int K, const float* W,
                         const float* q, const float* k, const float* b,
                         float* outh) {
        const int frags = RGAT_R * (K >> 5) * 4 * 32;     // pack threads
        const int waves = RGAT_R * (n_nodes / 16);        // 16x64 strip per wave
        const int gemmb = (waves * 32 + 255) / 256;
        init_buffers<<<nb64, 256, 0, stream>>>(mmax, ssum, acc, n_nodes);
        pack_weights<<<(frags + 255) / 256, 256, 0, stream>>>(W, Wp, K);
        rgat_gemm_wmma<<<gemmb, 256, 0, stream>>>(Act, Wp, T, n_nodes, K);
        qk_dot<<<qkb, 256, 0, stream>>>(T, q, k, dq, dk, RGAT_R * n_nodes);
        edge_logit_max<<<eb, 256, 0, stream>>>(src, dst, et, dq, dk, mmax, ex, E, n_nodes);
        edge_exp_sum<<<eb, 256, 0, stream>>>(dst, mmax, ex, ssum, E);
        edge_aggregate<<<ewb, 256, 0, stream>>>(src, dst, et, T, ex, ssum, acc, E, n_nodes);
        bias_relu<<<nb64, 256, 0, stream>>>(acc, b, outh, n_nodes);
    };

    float* out_mu = (float*)d_out;
    float* out_ls = out_mu + (size_t)n_nodes * RGAT_OUT;
    const int lb = (n_nodes * RGAT_OUT + 255) / 256;

    // Layer 1: x (K=128) -> hidden
    run_layer(x, 128, W1, q1, k1, b1, hid);
    // mu branch: hidden -> h2 -> linear
    run_layer(hid, 64, Wmu, qmu, kmu, bmu, h2);
    final_linear<<<lb, 256, 0, stream>>>(h2, Wm, bm, out_mu, 1.0f, n_nodes);
    // logvar branch: hidden -> h2 -> 0.5 * linear
    run_layer(hid, 64, Wlv, qlv, klv, blv, h2);
    final_linear<<<lb, 256, 0, stream>>>(h2, Wl, bl, out_ls, 0.5f, n_nodes);
}
